// _IPEXScaleDotProductRef_17179869184381
// MI455X (gfx1250) — compile-verified, matrix operand order checked by reading, not run on hardware
//
#include <hip/hip_runtime.h>

// ---------------------------------------------------------------------------
// Causal GQA attention (flash-attention-2 style) for gfx1250 / MI455X.
// B=2, H=32, HKV=8, T=2048, D=128, fp32 in/out, bf16 WMMA compute.
// bf16 tiles in LDS, double-buffered KV, register-staged global loads
// overlapped with WMMA, DPP ROW_XMASK butterfly softmax reductions
// (no ds_bpermute), k-permuted P/V^T layout for packed P stores.
// ---------------------------------------------------------------------------

typedef __bf16 bf16;
typedef bf16  v2bf  __attribute__((ext_vector_type(2)));
typedef bf16  v4bf  __attribute__((ext_vector_type(4)));
typedef bf16  v8bf  __attribute__((ext_vector_type(8)));
typedef bf16  v16bf __attribute__((ext_vector_type(16)));
typedef float v8f   __attribute__((ext_vector_type(8)));

#define B_   2
#define H_   32
#define HKV_ 8
#define T_   2048
#define D_   128
#define BQ   128   // query rows per workgroup (8 waves * 16)
#define BK   32    // kv rows per tile (WMMA K depth)
#define NWAVE 8
#define KPAD (D_ + 8)   // 136 bf16 -> 272B row stride (16B aligned, conflict-free)
#define VPAD (BK + 8)   // 40 bf16  -> 80B row stride
#define NEG_MASK (-1e9f)

template <bool V> struct BoolTag { static constexpr bool value = V; };

// Start k-index of the contiguous 8-element group g (0=lo,1=hi) for a 16-bit
// A/B WMMA fragment, half h = lane/16:  lo -> k=8h..8h+7, hi -> 16+8h..16+8h+7.
__device__ __forceinline__ int kgrp(int g, int h) { return g * 16 + 8 * h; }

__device__ __forceinline__ v8f wmma_bf16(v16bf a, v16bf b, v8f c) {
    return __builtin_amdgcn_wmma_f32_16x16x32_bf16(
        /*neg_a=*/false, a, /*neg_b=*/false, b,
        /*c_mod=*/(short)0, c, /*reuse_a=*/false, /*reuse_b=*/false);
}

__device__ __forceinline__ v16bf cat8(v8bf lo, v8bf hi) {
    return __builtin_shufflevector(lo, hi, 0, 1, 2, 3, 4, 5, 6, 7,
                                           8, 9, 10, 11, 12, 13, 14, 15);
}

// DPP16 ROW_XMASK butterfly move within each 16-lane row (pure VALU,
// co-executes with the WMMA/XDL stream; replaces ds_bpermute shuffles).
template <int XORM>
__device__ __forceinline__ float dpp_bfly(float v) {
    return __int_as_float(__builtin_amdgcn_update_dpp(
        0, __float_as_int(v), 0x160 | XORM, 0xf, 0xf, true));
}
__device__ __forceinline__ float row_max(float v) {
    v = fmaxf(v, dpp_bfly<1>(v));
    v = fmaxf(v, dpp_bfly<2>(v));
    v = fmaxf(v, dpp_bfly<4>(v));
    v = fmaxf(v, dpp_bfly<8>(v));
    return v;
}
__device__ __forceinline__ float row_sum(float v) {
    v += dpp_bfly<1>(v);
    v += dpp_bfly<2>(v);
    v += dpp_bfly<4>(v);
    v += dpp_bfly<8>(v);
    return v;
}

__global__ __launch_bounds__(256, 1)
void fa_causal_gqa_kernel(const float* __restrict__ Q,
                          const float* __restrict__ K,
                          const float* __restrict__ V,
                          const float* __restrict__ scale_attn,
                          float* __restrict__ out) {
    // Double-buffered bf16 tiles in LDS (~47 KB of the 320 KB WGP pool)
    __shared__ bf16 sKb [2][BK][KPAD];      // K tile   [s][d]
    __shared__ bf16 sVTb[2][D_][VPAD];      // V tile^T [d][s'], s'=2(s%16)+(s/16)
    __shared__ bf16 sP  [NWAVE][16][VPAD];  // per-wave P tile [m][s']

    const int tid  = threadIdx.x;
    const int w    = tid >> 5;          // wave id 0..7
    const int lane = tid & 31;
    const int col  = lane & 15;         // n / m position within 16
    const int h    = lane >> 4;         // lane half

    const int qt = blockIdx.x;          // q-tile of 128 rows
    const int hh = blockIdx.y;          // query head
    const int bb = blockIdx.z;

    const size_t bh   = (size_t)bb * H_ + hh;
    const size_t bhkv = (size_t)bb * HKV_ + (hh >> 2);   // n_rep = 4
    const float* __restrict__ Qb = Q + bh * T_ * D_;
    const float* __restrict__ Kb = K + bhkv * T_ * D_;
    const float* __restrict__ Vb = V + bhkv * T_ * D_;
    float*       __restrict__ Ob = out + bh * T_ * D_;

    const float inv_scale = 1.0f / scale_attn[0];

    const int qbase = qt * BQ + w * 16;
    const int qend  = qbase + 15;
    const int nkb   = qt * (BQ / BK) + (BQ / BK);  // causal tile count for WG

    // --- load Q fragments once (A-layout: lane row m = col).
    //     1/sqrt(D) is folded in here: (Q/scale)K^T == QK^T/scale.
    v16bf aQ[4];
    {
        const float* qrow = Qb + (size_t)(qbase + col) * D_;
        #pragma unroll
        for (int c4 = 0; c4 < 4; ++c4) {
            #pragma unroll
            for (int g = 0; g < 2; ++g) {
                const int k0 = c4 * 32 + kgrp(g, h);
                const float4 a = *(const float4*)(qrow + k0);
                const float4 b = *(const float4*)(qrow + k0 + 4);
                aQ[c4][g * 8 + 0] = (bf16)(a.x * inv_scale);
                aQ[c4][g * 8 + 1] = (bf16)(a.y * inv_scale);
                aQ[c4][g * 8 + 2] = (bf16)(a.z * inv_scale);
                aQ[c4][g * 8 + 3] = (bf16)(a.w * inv_scale);
                aQ[c4][g * 8 + 4] = (bf16)(b.x * inv_scale);
                aQ[c4][g * 8 + 5] = (bf16)(b.y * inv_scale);
                aQ[c4][g * 8 + 6] = (bf16)(b.z * inv_scale);
                aQ[c4][g * 8 + 7] = (bf16)(b.w * inv_scale);
            }
        }
    }

    // --- accumulators / softmax state
    v8f o[8] = {};
    float m_i[8], l_i[8];
    #pragma unroll
    for (int r = 0; r < 8; ++r) { m_i[r] = -1e30f; l_i[r] = 0.0f; }

    // ---- prologue: stage tile 0 into buffer 0
    {
        #pragma unroll
        for (int i = 0; i < 4; ++i) {
            const int f  = tid + i * 256;     // float4 index, 1024 total
            const int s  = f >> 5;
            const int dc = f & 31;
            const int sp = 2 * (s & 15) + (s >> 4);   // permuted k index
            const float4 k4 = *(const float4*)(Kb + (size_t)s * D_ + dc * 4);
            const float4 v4 = *(const float4*)(Vb + (size_t)s * D_ + dc * 4);
            v4bf kbv;
            kbv[0] = (bf16)k4.x; kbv[1] = (bf16)k4.y;
            kbv[2] = (bf16)k4.z; kbv[3] = (bf16)k4.w;
            *(v4bf*)&sKb[0][s][dc * 4] = kbv;
            sVTb[0][dc * 4 + 0][sp] = (bf16)v4.x;
            sVTb[0][dc * 4 + 1][sp] = (bf16)v4.y;
            sVTb[0][dc * 4 + 2][sp] = (bf16)v4.z;
            sVTb[0][dc * 4 + 3][sp] = (bf16)v4.w;
        }
    }
    __syncthreads();

    for (int kb = 0; kb < nkb; ++kb) {
        const int kvbase = kb * BK;
        const int cur    = kb & 1;
        const bool more  = (kb + 1 < nkb);

        // ---- issue next tile's global loads into registers (overlaps WMMA)
        float4 kreg[4], vreg[4];
        if (more) {
            #pragma unroll
            for (int i = 0; i < 4; ++i) {
                const int f  = tid + i * 256;
                const int s  = f >> 5;
                const int dc = f & 31;
                kreg[i] = *(const float4*)(Kb + (size_t)(kvbase + BK + s) * D_ + dc * 4);
                vreg[i] = *(const float4*)(Vb + (size_t)(kvbase + BK + s) * D_ + dc * 4);
            }
            if (kb + 2 < nkb) {   // gfx1250 global_prefetch_b8 for tile after next
                __builtin_prefetch(Kb + (size_t)(kvbase + 2 * BK + lane) * D_, 0, 1);
                __builtin_prefetch(Vb + (size_t)(kvbase + 2 * BK + lane) * D_, 0, 1);
            }
        }

        // ---- wave-uniform causal gating (EXEC stays all-ones for WMMA)
        if (kvbase <= qend) {
            // S = (Q/scale) * K^T : two 16-col subtiles, K-depth 128 in 4 chunks
            v8f s0 = {}, s1 = {};
            #pragma unroll
            for (int c4 = 0; c4 < 4; ++c4) {
                const v8bf a0 = *(const v8bf*)&sKb[cur][col]     [c4 * 32 + kgrp(0, h)];
                const v8bf a1 = *(const v8bf*)&sKb[cur][col]     [c4 * 32 + kgrp(1, h)];
                const v8bf b0 = *(const v8bf*)&sKb[cur][16 + col][c4 * 32 + kgrp(0, h)];
                const v8bf b1 = *(const v8bf*)&sKb[cur][16 + col][c4 * 32 + kgrp(1, h)];
                s0 = wmma_bf16(aQ[c4], cat8(a0, a1), s0);
                s1 = wmma_bf16(aQ[c4], cat8(b0, b1), s1);
            }

            // ---- online softmax; masking only on diagonal-intersecting tiles
            auto softmax_update = [&](auto masked_tag) {
                constexpr bool MASKED = decltype(masked_tag)::value;
                #pragma unroll
                for (int r = 0; r < 8; ++r) {
                    float x0 = s0[r];
                    float x1 = s1[r];
                    if (MASKED) {
                        const int qm = qbase + r + 8 * h;
                        x0 += ((kvbase + col)      > qm ? NEG_MASK : 0.0f);
                        x1 += ((kvbase + 16 + col) > qm ? NEG_MASK : 0.0f);
                    }
                    const float mx   = row_max(fmaxf(x0, x1));
                    const float mnew = fmaxf(m_i[r], mx);
                    const float p0 = __expf(x0 - mnew);
                    const float p1 = __expf(x1 - mnew);
                    const float rs = row_sum(p0 + p1);
                    const float alpha = __expf(m_i[r] - mnew);
                    l_i[r] = l_i[r] * alpha + rs;
                    m_i[r] = mnew;
                    #pragma unroll
                    for (int nt = 0; nt < 8; ++nt) o[nt][r] *= alpha;
                    // packed store: pair (n=col, n=col+16) -> s' = 2*col, 2*col+1
                    v2bf pk;
                    pk[0] = (bf16)p0; pk[1] = (bf16)p1;
                    *(v2bf*)&sP[w][r + 8 * h][2 * col] = pk;
                }
            };
            if (kvbase + BK - 1 <= qbase) softmax_update(BoolTag<false>{});
            else                          softmax_update(BoolTag<true>{});

            // ---- reload P in A-layout over permuted k (wave-private tile)
            const v8bf plo = *(const v8bf*)&sP[w][col][kgrp(0, h)];
            const v8bf phi = *(const v8bf*)&sP[w][col][kgrp(1, h)];
            const v16bf aP = cat8(plo, phi);

            // ---- O += P * V : 8 n-subtiles of 16 columns (same k-permutation)
            #pragma unroll
            for (int nt = 0; nt < 8; ++nt) {
                const int dcol = nt * 16 + col;
                const v8bf vlo = *(const v8bf*)&sVTb[cur][dcol][kgrp(0, h)];
                const v8bf vhi = *(const v8bf*)&sVTb[cur][dcol][kgrp(1, h)];
                o[nt] = wmma_bf16(aP, cat8(vlo, vhi), o[nt]);
            }
        }

        // ---- convert + store next tile into the other buffer
        if (more) {
            #pragma unroll
            for (int i = 0; i < 4; ++i) {
                const int f  = tid + i * 256;
                const int s  = f >> 5;
                const int dc = f & 31;
                const int sp = 2 * (s & 15) + (s >> 4);
                v4bf kbv;
                kbv[0] = (bf16)kreg[i].x; kbv[1] = (bf16)kreg[i].y;
                kbv[2] = (bf16)kreg[i].z; kbv[3] = (bf16)kreg[i].w;
                *(v4bf*)&sKb[cur ^ 1][s][dc * 4] = kbv;
                sVTb[cur ^ 1][dc * 4 + 0][sp] = (bf16)vreg[i].x;
                sVTb[cur ^ 1][dc * 4 + 1][sp] = (bf16)vreg[i].y;
                sVTb[cur ^ 1][dc * 4 + 2][sp] = (bf16)vreg[i].z;
                sVTb[cur ^ 1][dc * 4 + 3][sp] = (bf16)vreg[i].w;
            }
        }

        __syncthreads();   // next buffer staged; current reads complete
    }

    // ---- epilogue: O / l, write fp32 output
    #pragma unroll
    for (int r = 0; r < 8; ++r) {
        const int qm = qbase + r + 8 * h;
        const float invl = 1.0f / l_i[r];
        float* op = Ob + (size_t)qm * D_;
        #pragma unroll
        for (int nt = 0; nt < 8; ++nt)
            op[nt * 16 + col] = o[nt][r] * invl;
    }
}

extern "C" void kernel_launch(void* const* d_in, const int* in_sizes, int n_in,
                              void* d_out, int out_size, void* d_ws, size_t ws_size,
                              hipStream_t stream) {
    const float* Q     = (const float*)d_in[0];
    const float* Kp    = (const float*)d_in[1];
    const float* Vp    = (const float*)d_in[2];
    const float* scale = (const float*)d_in[3];
    // d_in[4] (additive causal mask) intentionally unused: reproduced
    // analytically in-kernel with the same -1e9 additive value.
    float* out = (float*)d_out;

    dim3 grid(T_ / BQ, H_, B_);   // (16, 32, 2) = 1024 workgroups
    dim3 block(256);              // 8 waves (wave32)
    fa_causal_gqa_kernel<<<grid, block, 0, stream>>>(Q, Kp, Vp, scale, out);
}
